// SelfAttention_7825430413840
// MI455X (gfx1250) — matrix operand order repaired and checked by reading, not compiled, not run on hardware
//
#include <hip/hip_runtime.h>
#include <hip/hip_bf16.h>
#include <stdint.h>

// ---------------------------------------------------------------------------
// Self-attention block for MI455X (gfx1250, wave32, WMMA).
// GEMMs + attention on v_wmma_f32_16x16x32_bf16 (fp32 accumulate).
// Global->LDS staging via global_load_async_to_lds_b128 (ASYNCcnt),
// double-buffered so tile i+1 streams in while tile i is computed.
// Flash attention computes S^T = K * Q^T so P converts C-layout -> A-layout
// entirely in registers (no LDS staging; 2 shuffles/step for softmax stats).
// Shapes hardcoded to the reference: B=4, S=4096, E=2048, H=16, hd=128.
// ---------------------------------------------------------------------------

typedef __attribute__((ext_vector_type(16))) __bf16 bf16x16;
typedef __attribute__((ext_vector_type(8)))  float  f32x8;

constexpr int Bc    = 4;
constexpr int Sc    = 4096;
constexpr int Ec    = 2048;
constexpr int Hc    = 16;
constexpr int HDc   = 128;            // head dim
constexpr int Mrows = Bc * Sc;        // 16384
constexpr int Nqkv  = 3 * Ec;         // 6144

union FragB16 {                       // 16 bf16 = 32 bytes per lane
  bf16x16 v;
  uint4   q[2];
  unsigned short us[16];
};

__device__ __forceinline__ unsigned short f2bf(float f) {
  unsigned int u = __float_as_uint(f);
  unsigned int r = u + 0x7FFFu + ((u >> 16) & 1u);   // round-to-nearest-even
  return (unsigned short)(r >> 16);
}
__device__ __forceinline__ float bf2f(unsigned short h) {
  return __uint_as_float(((unsigned int)h) << 16);
}

__device__ __forceinline__ f32x8 wmma_bf16(bf16x16 a, bf16x16 b, f32x8 c) {
  return __builtin_amdgcn_wmma_f32_16x16x32_bf16(
      false, a, false, b, (short)0, c, false, false);
}

// LDS offset of a __shared__ pointer (flat addr low 32 bits == LDS offset)
__device__ __forceinline__ unsigned lds_off(const void* p) {
  return (unsigned)(uintptr_t)p;
}
// Async global->LDS 16-byte copy, GVS mode: mem = saddr + voff(bytes)
__device__ __forceinline__ void async_b128(unsigned lds_dst, unsigned voff,
                                           const void* saddr) {
  asm volatile("global_load_async_to_lds_b128 %0, %1, %2"
               :: "v"(lds_dst), "v"(voff), "s"(saddr) : "memory");
}
__device__ __forceinline__ void wait_async0() {
  asm volatile("s_wait_asynccnt 0" ::: "memory");
}

// ---------------------------------------------------------------------------
// Kernel 1: fp32 -> bf16 conversion (n divisible by 4)
// ---------------------------------------------------------------------------
__global__ void cvt_bf16_kernel(const float* __restrict__ src,
                                unsigned short* __restrict__ dst, int n) {
  int i = (blockIdx.x * blockDim.x + threadIdx.x) << 2;
  if (i >= n) return;
  float4 f = *(const float4*)(src + i);
  uint2 o;
  o.x = (unsigned)f2bf(f.x) | ((unsigned)f2bf(f.y) << 16);
  o.y = (unsigned)f2bf(f.z) | ((unsigned)f2bf(f.w) << 16);
  *(uint2*)(dst + i) = o;
}

// ---------------------------------------------------------------------------
// Kernel 2: LayerNorm over E=2048, one block (256 threads) per row, bf16 out
// ---------------------------------------------------------------------------
__global__ __launch_bounds__(256)
void ln_bf16_kernel(const float* __restrict__ x,
                    const float* __restrict__ lnw,
                    const float* __restrict__ lnb,
                    unsigned short* __restrict__ xn) {
  const int row = blockIdx.x;
  const float* xr = x + (size_t)row * Ec;
  float v[8];
  float s = 0.f, s2 = 0.f;
#pragma unroll
  for (int i = 0; i < 8; ++i) {
    v[i] = xr[threadIdx.x + (i << 8)];
    s += v[i]; s2 += v[i] * v[i];
  }
#pragma unroll
  for (int off = 16; off > 0; off >>= 1) {
    s  += __shfl_xor(s,  off, 32);
    s2 += __shfl_xor(s2, off, 32);
  }
  __shared__ float rs[8], rs2[8];
  const int wv = threadIdx.x >> 5;
  if ((threadIdx.x & 31) == 0) { rs[wv] = s; rs2[wv] = s2; }
  __syncthreads();
  s = 0.f; s2 = 0.f;
#pragma unroll
  for (int i = 0; i < 8; ++i) { s += rs[i]; s2 += rs2[i]; }
  const float mean = s  * (1.f / Ec);
  const float var  = s2 * (1.f / Ec) - mean * mean;
  const float rstd = rsqrtf(var + 1e-5f);
  unsigned short* xo = xn + (size_t)row * Ec;
#pragma unroll
  for (int i = 0; i < 8; ++i) {
    const int c = threadIdx.x + (i << 8);
    xo[c] = f2bf((v[i] - mean) * rstd * lnw[c] + lnb[c]);
  }
}

// ---------------------------------------------------------------------------
// Kernel 3/6: C(MxN) = A_bf16(MxK) * W_bf16(NxK)^T + bias, WMMA tiled.
// Block 256 threads (8 waves); block tile 256x64 (wave tile 32x64), BK=32;
// double-buffered async LDS staging.
// ---------------------------------------------------------------------------
template <bool OUT_BF16>
__global__ __launch_bounds__(256)
void gemm_wmma_kernel(const unsigned short* __restrict__ A,
                      const unsigned short* __restrict__ W,
                      const float* __restrict__ bias,
                      void* __restrict__ out,
                      int M, int N, int K) {
  __shared__ unsigned short As[2][256 * 32];   // 2 x 16 KB
  __shared__ unsigned short Ws[2][64 * 32];    // 2 x 4 KB
  const int m0   = blockIdx.x * 256;
  const int n0   = blockIdx.y * 64;
  const int t    = threadIdx.x;
  const int wave = t >> 5, lane = t & 31;
  const int hi   = (lane & 16) ? 1 : 0;
  const int l15  = lane & 15;

  auto issue = [&](int k0, int buf) {
    const unsigned ldsA = lds_off(&As[buf][0]);
    const unsigned ldsW = lds_off(&Ws[buf][0]);
    // A tile 256x32: 1024 x 16B, 4 async per thread
#pragma unroll
    for (int i = 0; i < 4; ++i) {
      int idx = t + (i << 8);
      int row = idx >> 2, ch = idx & 3;
      async_b128(ldsA + (unsigned)(row * 32 + ch * 8) * 2u,
                 (unsigned)(((size_t)(m0 + row) * K + k0 + ch * 8) * 2u), A);
    }
    { // W tile 64x32: 256 x 16B, 1 async per thread
      int row = t >> 2, ch = t & 3;
      async_b128(ldsW + (unsigned)(row * 32 + ch * 8) * 2u,
                 (unsigned)(((size_t)(n0 + row) * K + k0 + ch * 8) * 2u), W);
    }
  };

  f32x8 acc[2][4];
#pragma unroll
  for (int mr = 0; mr < 2; ++mr)
#pragma unroll
    for (int g = 0; g < 4; ++g)
#pragma unroll
      for (int r = 0; r < 8; ++r) acc[mr][g][r] = 0.f;

  issue(0, 0);
  for (int k0 = 0; k0 < K; k0 += 32) {
    const int buf = (k0 >> 5) & 1;
    wait_async0();                 // current tile's async copies done
    __syncthreads();               // visible to all waves; prev buf free
    if (k0 + 32 < K) issue(k0 + 32, buf ^ 1);

    FragB16 a0, a1;
    const int ar = wave * 32 + l15;
    a0.q[0] = *(const uint4*)&As[buf][ar * 32 + hi * 8];
    a0.q[1] = *(const uint4*)&As[buf][ar * 32 + 16 + hi * 8];
    a1.q[0] = *(const uint4*)&As[buf][(ar + 16) * 32 + hi * 8];
    a1.q[1] = *(const uint4*)&As[buf][(ar + 16) * 32 + 16 + hi * 8];
#pragma unroll
    for (int g = 0; g < 4; ++g) {
      FragB16 b;
      const int wr = g * 16 + l15;
      b.q[0] = *(const uint4*)&Ws[buf][wr * 32 + hi * 8];
      b.q[1] = *(const uint4*)&Ws[buf][wr * 32 + 16 + hi * 8];
      acc[0][g] = wmma_bf16(a0.v, b.v, acc[0][g]);
      acc[1][g] = wmma_bf16(a1.v, b.v, acc[1][g]);
    }
  }

#pragma unroll
  for (int mr = 0; mr < 2; ++mr)
#pragma unroll
    for (int g = 0; g < 4; ++g) {
      const int n = n0 + g * 16 + l15;
      const float bn = bias[n];
#pragma unroll
      for (int r = 0; r < 8; ++r) {
        const int m = m0 + wave * 32 + mr * 16 + r + hi * 8;
        const float val = acc[mr][g][r] + bn;
        if (OUT_BF16) ((unsigned short*)out)[(size_t)m * N + n] = f2bf(val);
        else          ((float*)out)[(size_t)m * N + n] = val;
      }
    }
}

// ---------------------------------------------------------------------------
// Kernel 4: RoPE in-place on bf16 q,k; attention scale folded into q.
// ---------------------------------------------------------------------------
__global__ __launch_bounds__(256)
void rope_kernel(unsigned short* __restrict__ qkv,
                 const float* __restrict__ freqs,
                 const float* __restrict__ pos) {
  const int t = blockIdx.x * 256 + threadIdx.x;   // 2^24 threads
  const int j = t & 63;
  const int h = (t >> 6) & 15;
  const int s = (t >> 10) & (Sc - 1);
  const int b = t >> 22;
  const float ang = freqs[j * 3 + 0] * pos[s * 3 + 0] +
                    freqs[j * 3 + 1] * pos[s * 3 + 1] +
                    freqs[j * 3 + 2] * pos[s * 3 + 2];
  const float cs = __cosf(ang), sn = __sinf(ang);
  const float qscale = 0.08838834764831845f;      // 1/sqrt(128)
  const size_t base = ((size_t)(b * Sc + s)) * Nqkv + h * HDc + 2 * j;
#pragma unroll
  for (int which = 0; which < 2; ++which) {       // 0 = q (scaled), 1 = k
    unsigned short* p = qkv + base + which * Ec;
    unsigned u = *(const unsigned*)p;
    float tr = bf2f((unsigned short)(u & 0xffff));
    float ti = bf2f((unsigned short)(u >> 16));
    float nr = tr * cs - ti * sn;
    float ni = tr * sn + ti * cs;
    if (which == 0) { nr *= qscale; ni *= qscale; }
    *(unsigned*)p = (unsigned)f2bf(nr) | ((unsigned)f2bf(ni) << 16);
  }
}

// ---------------------------------------------------------------------------
// Kernel 5: flash attention, S^T formulation, double-buffered K/V tiles.
// Block = 128 threads (4 waves), 64 query rows; 32 keys per iteration.
// ---------------------------------------------------------------------------
__global__ __launch_bounds__(128)
void flash_attn_kernel(const unsigned short* __restrict__ qkv,
                       unsigned short* __restrict__ attn) {
  __shared__ unsigned short Kt[2][32 * HDc];   // 2 x 8 KB  [key][d]
  __shared__ unsigned short Vt[2][HDc * 32];   // 2 x 8 KB  [d][key]

  const int qtile = blockIdx.x * 64;
  const int h     = blockIdx.y;
  const int b     = blockIdx.z;
  const int t     = threadIdx.x;
  const int wave  = t >> 5, lane = t & 31;
  const int hi    = (lane & 16) ? 1 : 0;
  const int l15   = lane & 15;

  const size_t ld = (size_t)Nqkv;              // qkv row stride (elements)
  const unsigned short* qp = qkv + (size_t)b * Sc * ld + (size_t)h * HDc;
  const unsigned short* kp = qp + Ec;
  const unsigned short* vp = qp + 2 * Ec;

  auto issue = [&](int kt, int buf) {
    const unsigned ldsK = lds_off(&Kt[buf][0]);
    // K tile 32x128 row-major: 512 x 16B async copies, 4 per thread
#pragma unroll
    for (int i = 0; i < 4; ++i) {
      int idx = t + (i << 7);
      int row = idx >> 4, ch = idx & 15;
      async_b128(ldsK + (unsigned)(row * HDc + ch * 8) * 2u,
                 (unsigned)(((size_t)(kt * 32 + row) * ld + ch * 8) * 2u), kp);
    }
    // V tile transposed: thread owns key-pair, 16 d; packed b32 stores
    int pr = t >> 3;                 // key pair 0..15
    int dseg = (t & 7) << 4;         // d segment start
    const unsigned short* r0 = vp + (size_t)(kt * 32 + 2 * pr) * ld + dseg;
    const unsigned short* r1 = r0 + ld;
    unsigned* vt32 = (unsigned*)&Vt[buf][0];
#pragma unroll
    for (int i = 0; i < 2; ++i) {
      uint4 a = *(const uint4*)(r0 + i * 8);
      uint4 bq = *(const uint4*)(r1 + i * 8);
      const unsigned aw[4] = {a.x, a.y, a.z, a.w};
      const unsigned bw[4] = {bq.x, bq.y, bq.z, bq.w};
#pragma unroll
      for (int j = 0; j < 4; ++j) {
        int d0 = dseg + i * 8 + 2 * j;
        vt32[d0 * 16 + pr]       = (aw[j] & 0xffffu) | (bw[j] << 16);
        vt32[(d0 + 1) * 16 + pr] = (aw[j] >> 16) | (bw[j] & 0xffff0000u);
      }
    }
  };

  // Q^T as B-fragments: lane = query column, contiguous d per chunk-half
  FragB16 qb[4];
  {
    const unsigned short* qr = qp + (size_t)(qtile + wave * 16 + l15) * ld;
#pragma unroll
    for (int c = 0; c < 4; ++c) {
      qb[c].q[0] = *(const uint4*)(qr + c * 32 + hi * 8);
      qb[c].q[1] = *(const uint4*)(qr + c * 32 + 16 + hi * 8);
    }
  }

  f32x8 o[8];                                  // O accumulator 16q x 128d
#pragma unroll
  for (int c = 0; c < 8; ++c)
#pragma unroll
    for (int r = 0; r < 8; ++r) o[c][r] = 0.f;
  float m1 = -1e30f, l1 = 0.f;                 // per-lane (query) stats

  issue(0, 0);
  for (int kt = 0; kt < Sc / 32; ++kt) {
    const int buf = kt & 1;
    wait_async0();                 // current K tile resident
    __syncthreads();               // V tile + cross-wave visibility
    if (kt + 1 < Sc / 32) issue(kt + 1, buf ^ 1);

    // --- S^T = K * Q^T : s0 = keys 0-15, s1 = keys 16-31 (M), queries (N)
    f32x8 s0, s1;
#pragma unroll
    for (int r = 0; r < 8; ++r) { s0[r] = 0.f; s1[r] = 0.f; }
#pragma unroll
    for (int c = 0; c < 4; ++c) {
      FragB16 ka, kb;
      int off0 = l15 * HDc + c * 32;
      ka.q[0] = *(const uint4*)&Kt[buf][off0 + hi * 8];
      ka.q[1] = *(const uint4*)&Kt[buf][off0 + 16 + hi * 8];
      int off1 = (16 + l15) * HDc + c * 32;
      kb.q[0] = *(const uint4*)&Kt[buf][off1 + hi * 8];
      kb.q[1] = *(const uint4*)&Kt[buf][off1 + 16 + hi * 8];
      s0 = wmma_bf16(ka.v, qb[c].v, s0);
      s1 = wmma_bf16(kb.v, qb[c].v, s1);
    }

    // --- online softmax: per-lane (query) + one xor-16 exchange per stat
    float mx = fmaxf(s0[0], s1[0]);
#pragma unroll
    for (int r = 1; r < 8; ++r) mx = fmaxf(mx, fmaxf(s0[r], s1[r]));
    mx = fmaxf(mx, __shfl_xor(mx, 16, 32));
    const float nm = fmaxf(m1, mx);
    const float al = __expf(m1 - nm);
    m1 = nm;
    FragB16 pf;                                  // P as A-fragment, in-register
    float rsum = 0.f;
#pragma unroll
    for (int r = 0; r < 8; ++r) {
      float p0 = __expf(s0[r] - nm);
      float p1 = __expf(s1[r] - nm);
      rsum += p0 + p1;
      pf.us[r]     = f2bf(p0);                   // k = r + 8*hi
      pf.us[8 + r] = f2bf(p1);                   // k = 16 + r + 8*hi
    }
    rsum += __shfl_xor(rsum, 16, 32);
    l1 = l1 * al + rsum;

    // broadcast alpha to O rows (O C-layout rows = queries r + 8*hi)
    float alb[8];
#pragma unroll
    for (int r = 0; r < 8; ++r) alb[r] = __shfl(al, r + hi * 8, 32);
#pragma unroll
    for (int c = 0; c < 8; ++c)
#pragma unroll
      for (int r = 0; r < 8; ++r) o[c][r] *= alb[r];

    // --- O += P * V : 8 hd-chunks, B frags from transposed Vt
#pragma unroll
    for (int c = 0; c < 8; ++c) {
      FragB16 vb;
      int off = (c * 16 + l15) * 32 + hi * 8;
      vb.q[0] = *(const uint4*)&Vt[buf][off];
      vb.q[1] = *(const uint4*)&Vt[buf][off + 16];
      o[c] = wmma_bf16(pf.v, vb.v, o[c]);
    }
  }

  // normalize (broadcast 1/l to O rows) and write attn bf16
  float li = 1.f / l1;
  float lb[8];
#pragma unroll
  for (int r = 0; r < 8; ++r) lb[r] = __shfl(li, r + hi * 8, 32);
#pragma unroll
  for (int c = 0; c < 8; ++c)
#pragma unroll
    for (int r = 0; r < 8; ++r) {
      int m = qtile + wave * 16 + r + hi * 8;
      size_t idx = ((size_t)b * Sc + m) * Ec + h * HDc + c * 16 + l15;
      attn[idx] = f2bf(o[c][r] * lb[r]);
    }
}

// ---------------------------------------------------------------------------
// Launch
// ---------------------------------------------------------------------------
extern "C" void kernel_launch(void* const* d_in, const int* in_sizes, int n_in,
                              void* d_out, int out_size, void* d_ws, size_t ws_size,
                              hipStream_t stream) {
  (void)in_sizes; (void)n_in; (void)out_size; (void)ws_size;
  const float* x    = (const float*)d_in[0];
  const float* pos  = (const float*)d_in[1];
  const float* lnw  = (const float*)d_in[2];
  const float* lnb  = (const float*)d_in[3];
  const float* qkvw = (const float*)d_in[4];
  const float* qkvb = (const float*)d_in[5];
  const float* outw = (const float*)d_in[6];
  const float* outb = (const float*)d_in[7];
  const float* frq  = (const float*)d_in[8];

  char* ws = (char*)d_ws;
  size_t off = 0;
  auto take = [&](size_t bytes) {
    char* p = ws + off;
    off += (bytes + 255) & ~(size_t)255;
    return p;
  };
  unsigned short* xn_bf   = (unsigned short*)take((size_t)Mrows * Ec * 2);
  unsigned short* qkv_bf  = (unsigned short*)take((size_t)Mrows * Nqkv * 2);
  unsigned short* attn_bf = (unsigned short*)take((size_t)Mrows * Ec * 2);
  unsigned short* qkvw_bf = (unsigned short*)take((size_t)Nqkv * Ec * 2);
  unsigned short* outw_bf = (unsigned short*)take((size_t)Ec * Ec * 2);

  cvt_bf16_kernel<<<(Nqkv * Ec / 4 + 255) / 256, 256, 0, stream>>>(qkvw, qkvw_bf, Nqkv * Ec);
  cvt_bf16_kernel<<<(Ec * Ec / 4 + 255) / 256, 256, 0, stream>>>(outw, outw_bf, Ec * Ec);
  ln_bf16_kernel<<<Mrows, 256, 0, stream>>>(x, lnw, lnb, xn_bf);
  gemm_wmma_kernel<true><<<dim3(Mrows / 256, Nqkv / 64), 256, 0, stream>>>(
      xn_bf, qkvw_bf, qkvb, qkv_bf, Mrows, Nqkv, Ec);
  rope_kernel<<<(Bc * Sc * Hc * (HDc / 2)) / 256, 256, 0, stream>>>(qkv_bf, frq, pos);
  flash_attn_kernel<<<dim3(Sc / 64, Hc, Bc), 128, 0, stream>>>(qkv_bf, attn_bf);
  gemm_wmma_kernel<false><<<dim3(Mrows / 256, Ec / 64), 256, 0, stream>>>(
      attn_bf, outw_bf, outb, d_out, Mrows, Ec, Ec);
}
